// selfAttn_merge_SRNN_77850577207862
// MI455X (gfx1250) — matrix operand order repaired and checked by reading, not compiled, not run on hardware
//
#include <hip/hip_runtime.h>

// dims
#define S_  128
#define E_  256
#define H_  20
#define D_  64
#define A_  64
#define NH_ 8
#define PT  16          // (s,e) pairs per tile
#define NBLK 512        // persistent blocks
#define TPB  4          // tiles per block (512*4*16 = 32768 pairs)
#define NEGV (-1e9f)

typedef __attribute__((ext_vector_type(16))) __bf16          v16bf;
typedef __attribute__((ext_vector_type(16))) unsigned short  v16us;
typedef __attribute__((ext_vector_type(8)))  float           v8f;
typedef __attribute__((ext_vector_type(4)))  unsigned int    u32x4;
typedef __attribute__((ext_vector_type(8)))  int             i32x8;
typedef __attribute__((ext_vector_type(4)))  int             i32x4;

union FragBF { v16us u; v16bf b; };

static __device__ __forceinline__ unsigned short f2bf(float f) {
    unsigned u = __float_as_uint(f);
    unsigned r = u + 0x7FFFu + ((u >> 16) & 1u);   // round-to-nearest-even
    return (unsigned short)(r >> 16);
}
// A-matrix (16-bit, 16x32) per-lane K index for element e (ISA 7.12.2)
static __device__ __forceinline__ int a_k(int e, int hi) {
    int v = e >> 1, bit = e & 1;
    return ((v < 4) ? 0 : 16) + hi * 8 + 2 * (v & 3) + bit;
}
// B-matrix (16-bit, 32x16) per-lane K index for element e (same order as A)
static __device__ __forceinline__ int b_k(int e, int hi) {
    int v = e >> 1, bit = e & 1;
    return ((v < 4) ? 0 : 16) + hi * 8 + 2 * (v & 3) + bit;
}

// TDM: issue a 1-row copy of `nelts` f32 from `gsrc` into LDS offset `ldsoff`.
// D# per CDNA5 ISA.
static __device__ __forceinline__ void tdm_load_f32(unsigned ldsoff,
                                                    const float* gsrc,
                                                    unsigned nelts) {
    unsigned long long ga = (unsigned long long)(size_t)gsrc;
    u32x4 g0;
    g0[0] = 1u;                                        // count=1, user mode
    g0[1] = ldsoff;                                    // lds_addr (bytes)
    g0[2] = (unsigned)(ga & 0xFFFFFFFFu);              // global_addr[31:0]
    g0[3] = (unsigned)((ga >> 32) & 0x01FFFFFFu)       // global_addr[56:32]
          | (2u << 30);                                // type = 2 (image)
    i32x8 g1;
    g1[0] = (int)(2u << 16);                           // data_size = 2 -> 4 bytes
    g1[1] = (int)((nelts & 0xFFFFu) << 16);            // tensor_dim0[15:0]
    g1[2] = (int)(((nelts >> 16) & 0xFFFFu) | (1u << 16)); // td0[31:16], tensor_dim1=1
    g1[3] = (int)((nelts & 0xFFFFu) << 16);            // tile_dim0 (fits 16b: 20480)
    g1[4] = 1;                                         // tile_dim1 = 1
    g1[5] = (int)nelts;                                // tensor_dim0_stride[31:0]
    g1[6] = 0;
    g1[7] = 0;
    i32x4 z4 = {0, 0, 0, 0};
#if defined(__clang_major__) && (__clang_major__ >= 23)
    i32x8 z8 = {0, 0, 0, 0, 0, 0, 0, 0};
    __builtin_amdgcn_tensor_load_to_lds(g0, g1, z4, z4, z8, 0);
#else
    __builtin_amdgcn_tensor_load_to_lds(g0, g1, z4, z4, 0);
#endif
}

__global__ __launch_bounds__(256, 1)
void attn_merge_srnn_kernel(const float* __restrict__ hT, const float* __restrict__ hS,
                            const int*   __restrict__ seqlen,
                            const float* __restrict__ Wt, const float* __restrict__ bt,
                            const float* __restrict__ Ws, const float* __restrict__ bs,
                            const float* __restrict__ Wf, const float* __restrict__ bfv,
                            float* __restrict__ out)
{
    __shared__ float          sHf[2][PT * H_ * D_];   // double-buffered spatial tile (f32, TDM)
    __shared__ unsigned short sT[PT * D_];            // temporal tile, bf16
    __shared__ float          sScr[NH_ * PT * D_];    // per-wave te -> u scratch (f32)
    __shared__ float          sAttn[NH_ * PT * H_];   // attention weights
    __shared__ float          sOut[PT * D_];          // merged output accumulator
    __shared__ int            sLen[PT];

    const int tid  = threadIdx.x;
    const int wid  = tid >> 5;
    const int lane = tid & 31;
    const int n    = wid;                 // one wave per head
    const int row  = lane & 15;
    const int hi   = (lane >> 4) & 1;
    float* scr = sScr + n * (PT * D_);

    // ---- hoist: build all weight B-fragments once (L2-resident, reused 4 tiles) ----
    FragBF wtB[4][2], wsB[4][2], wfB[4][2];
    float  btv[4];
#pragma unroll
    for (int nt = 0; nt < 4; ++nt) {
        int col = nt * 16 + row;
        btv[nt] = bt[n * A_ + col];
#pragma unroll
        for (int kc = 0; kc < 2; ++kc) {
#pragma unroll
            for (int e = 0; e < 16; e += 2) {
                int k = kc * 32 + b_k(e, hi);
                wtB[nt][kc].u[e]     = f2bf(Wt[(size_t)(n * D_ + k)     * A_ + col]);
                wtB[nt][kc].u[e + 1] = f2bf(Wt[(size_t)(n * D_ + k + 1) * A_ + col]);
                const float2 wp = *(const float2*)(Ws + (size_t)(n * D_ + col) * A_ + k);
                wsB[nt][kc].u[e]     = f2bf(wp.x);    // Ws^T fragment
                wsB[nt][kc].u[e + 1] = f2bf(wp.y);
                wfB[nt][kc].u[e]     = f2bf(Wf[(size_t)(n * D_ + k)     * D_ + col]);
                wfB[nt][kc].u[e + 1] = f2bf(Wf[(size_t)(n * D_ + k + 1) * D_ + col]);
            }
        }
    }

    // ---- TDM: prime the pipeline with tile 0 of this block ----
    const int t0 = blockIdx.x * TPB;
    if (wid == 0)
        tdm_load_f32((unsigned)(size_t)&sHf[0][0],
                     hS + (size_t)(t0 * PT) * (H_ * D_), PT * H_ * D_);

    for (int it = 0; it < TPB; ++it) {
        const int t  = t0 + it;
        const int p0 = t * PT;
        const float* sHcur = &sHf[it & 1][0];

        // ---- stage A: small per-tile loads (regular path) ----
        {   // temporal: 16*64 floats = 256 float4 -> bf16
            float4 v = ((const float4*)(hT + (size_t)p0 * D_))[tid];
            unsigned* d32 = (unsigned*)sT;
            d32[tid * 2 + 0] = (unsigned)f2bf(v.x) | ((unsigned)f2bf(v.y) << 16);
            d32[tid * 2 + 1] = (unsigned)f2bf(v.z) | ((unsigned)f2bf(v.w) << 16);
        }
        for (int i = tid; i < PT * D_; i += 256) sOut[i] = 0.f;
        if (tid < PT) sLen[tid] = seqlen[p0 + tid];

        // ---- TDM: issue next tile, then wait for current ----
        if (wid == 0) {
            if (it + 1 < TPB) {
                tdm_load_f32((unsigned)(size_t)&sHf[(it + 1) & 1][0],
                             hS + (size_t)((t + 1) * PT) * (H_ * D_), PT * H_ * D_);
                __builtin_amdgcn_s_wait_tensorcnt((short)1);   // current tile done
            } else {
                __builtin_amdgcn_s_wait_tensorcnt((short)0);
            }
        }
        __syncthreads();

        // ---- A fragments of temporal tile (M=pair, K=d) ----
        FragBF aT[2];
#pragma unroll
        for (int kc = 0; kc < 2; ++kc)
#pragma unroll
            for (int e = 0; e < 16; ++e)
                aT[kc].u[e] = sT[row * D_ + kc * 32 + a_k(e, hi)];

        // ---- te = T @ Wt[n] + bt[n] (WMMA) -> scr[p][a] ----
#pragma unroll
        for (int nt = 0; nt < 4; ++nt) {
            v8f c;
#pragma unroll
            for (int j = 0; j < 8; ++j) c[j] = btv[nt];
            c = __builtin_amdgcn_wmma_f32_16x16x32_bf16(false, aT[0].b, false, wtB[nt][0].b,
                                                        (short)0, c, false, false);
            c = __builtin_amdgcn_wmma_f32_16x16x32_bf16(false, aT[1].b, false, wtB[nt][1].b,
                                                        (short)0, c, false, false);
#pragma unroll
            for (int v = 0; v < 8; ++v)
                scr[(v + 8 * hi) * D_ + nt * 16 + row] = c[v];
        }

        // ---- cP[p] = te[p,:] . bs[n,:]  (bias of the algebraic rewrite) ----
        float cP;
        {
            int p = lane >> 1, half = lane & 1;
            float acc = 0.f;
            for (int a = half * 32; a < half * 32 + 32; ++a)
                acc += scr[p * D_ + a] * bs[n * A_ + a];
            acc += __shfl_xor(acc, 1, 32);
            cP = acc;
        }

        // ---- A fragments of te (bf16) ----
        FragBF aTe[2];
#pragma unroll
        for (int kc = 0; kc < 2; ++kc)
#pragma unroll
            for (int e = 0; e < 16; ++e)
                aTe[kc].u[e] = f2bf(scr[row * D_ + kc * 32 + a_k(e, hi)]);

        // ---- u = te @ Ws[n]^T (WMMA) -> overwrite scr[p][d] ----
#pragma unroll
        for (int nt = 0; nt < 4; ++nt) {
            v8f c;
#pragma unroll
            for (int j = 0; j < 8; ++j) c[j] = 0.f;
            c = __builtin_amdgcn_wmma_f32_16x16x32_bf16(false, aTe[0].b, false, wsB[nt][0].b,
                                                        (short)0, c, false, false);
            c = __builtin_amdgcn_wmma_f32_16x16x32_bf16(false, aTe[1].b, false, wsB[nt][1].b,
                                                        (short)0, c, false, false);
#pragma unroll
            for (int v = 0; v < 8; ++v)
                scr[(v + 8 * hi) * D_ + nt * 16 + row] = c[v];
        }

        // ---- scores + masked softmax (VALU; 2 lanes per pair, 10 m's each) ----
        {
            int p = lane >> 1, half = lane & 1;
            int len = sLen[p];
            float sc[10];
            float mx = -3.0e38f;
            const float* urow = scr + p * D_;
#pragma unroll
            for (int j = 0; j < 10; ++j) {
                int m = half * 10 + j;
                float acc = cP;
                const float* hrow = sHcur + (p * H_ + m) * D_;
                for (int d = 0; d < D_; ++d) acc += hrow[d] * urow[d];
                sc[j] = (m < len) ? acc * 2.5f : NEGV;   // temperature = H/sqrt(A)
                mx = fmaxf(mx, sc[j]);
            }
            mx = fmaxf(mx, __shfl_xor(mx, 1, 32));
            float se = 0.f;
#pragma unroll
            for (int j = 0; j < 10; ++j) { sc[j] = __expf(sc[j] - mx); se += sc[j]; }
            se += __shfl_xor(se, 1, 32);
            float inv = 1.f / se;
#pragma unroll
            for (int j = 0; j < 10; ++j)
                sAttn[(n * PT + p) * H_ + half * 10 + j] = sc[j] * inv;
        }

        // ---- wv[p,d] = sum_m attn[p,m]*Hs[p,m,d], built directly as A fragments ----
        FragBF aWv[2];
        {
            float at[H_];
#pragma unroll
            for (int m = 0; m < H_; ++m) at[m] = sAttn[(n * PT + row) * H_ + m];
#pragma unroll
            for (int kc = 0; kc < 2; ++kc)
#pragma unroll
                for (int e = 0; e < 16; e += 2) {
                    int d = kc * 32 + a_k(e, hi);   // even; element e+1 is d+1
                    float acc0 = 0.f, acc1 = 0.f;
                    for (int m = 0; m < H_; ++m) {
                        const float2 hv = *(const float2*)(sHcur + (row * H_ + m) * D_ + d);
                        acc0 += at[m] * hv.x;
                        acc1 += at[m] * hv.y;
                    }
                    aWv[kc].u[e]     = f2bf(acc0);
                    aWv[kc].u[e + 1] = f2bf(acc1);
                }
        }

        // ---- final merge: out += wv_n @ Wf[n*64:(n+1)*64, :] (WMMA + ds_add_f32) ----
#pragma unroll
        for (int nt = 0; nt < 4; ++nt) {
            v8f c;
#pragma unroll
            for (int j = 0; j < 8; ++j) c[j] = 0.f;
            c = __builtin_amdgcn_wmma_f32_16x16x32_bf16(false, aWv[0].b, false, wfB[nt][0].b,
                                                        (short)0, c, false, false);
            c = __builtin_amdgcn_wmma_f32_16x16x32_bf16(false, aWv[1].b, false, wfB[nt][1].b,
                                                        (short)0, c, false, false);
#pragma unroll
            for (int v = 0; v < 8; ++v)
                atomicAdd(&sOut[(v + 8 * hi) * D_ + nt * 16 + row], c[v]);
        }
        __syncthreads();

        // ---- write out: out[p,:] = sOut[p,:] + bf ----
        for (int i = tid; i < PT * D_; i += 256) {
            int p = i >> 6, d = i & 63;
            out[(size_t)(p0 + p) * D_ + d] = sOut[i] + bfv[d];
        }
        __syncthreads();   // protect sT/sOut/next-buffer reuse
    }
}

extern "C" void kernel_launch(void* const* d_in, const int* in_sizes, int n_in,
                              void* d_out, int out_size, void* d_ws, size_t ws_size,
                              hipStream_t stream) {
    const float* hT  = (const float*)d_in[0];
    const float* hS  = (const float*)d_in[1];
    const int*   len = (const int*)  d_in[2];
    const float* Wt  = (const float*)d_in[3];
    const float* bt  = (const float*)d_in[4];
    const float* Ws  = (const float*)d_in[5];
    const float* bs  = (const float*)d_in[6];
    const float* Wf  = (const float*)d_in[7];
    const float* bf_ = (const float*)d_in[8];
    float* out = (float*)d_out;

    dim3 grid(NBLK);    // persistent: 512 blocks x 4 tiles, double-buffered TDM
    dim3 block(256);    // 8 waves -> 8 heads
    hipLaunchKernelGGL(attn_merge_srnn_kernel, grid, block, 0, stream,
                       hT, hS, len, Wt, bt, Ws, bs, Wf, bf_, out);
}
